// SupGCL_GConv_70866960384012
// MI455X (gfx1250) — matrix operand order repaired and checked by reading, not compiled, not run on hardware
//
#include <hip/hip_runtime.h>
#include <hip/hip_bf16.h>

typedef __attribute__((ext_vector_type(16))) _Float16 v16h;
typedef __attribute__((ext_vector_type(8)))  float    v8f;

#define FDIM 64
#define TILES_PER_WAVE 4

// ---------------------------------------------------------------------------
// init: deg = cnt = 1.0 (self loops), msg = 0
// ---------------------------------------------------------------------------
__global__ void init_kernel(float* __restrict__ deg, float* __restrict__ cnt,
                            float* __restrict__ msg, int n) {
    long long i = (long long)blockIdx.x * blockDim.x + threadIdx.x;
    long long tot = (long long)n * FDIM;
    if (i < tot) msg[i] = 0.0f;
    if (i < n) { deg[i] = 1.0f; cnt[i] = 1.0f; }
}

// ---------------------------------------------------------------------------
// degree histograms: deg over src, cnt over dst
// ---------------------------------------------------------------------------
__global__ void deg_cnt_kernel(const int* __restrict__ src, const int* __restrict__ dst,
                               float* __restrict__ deg, float* __restrict__ cnt, int E) {
    int e = blockIdx.x * blockDim.x + threadIdx.x;
    if (e >= E) return;
    atomicAdd(&deg[src[e]], 1.0f);
    atomicAdd(&cnt[dst[e]], 1.0f);
}

// ---------------------------------------------------------------------------
// WMMA GEMM: out = relu(X[n x 64] @ W[64 x 64] + b)
// 128 threads = 4 waves. W is pre-swizzled in LDS into this kernel's contiguous-k
// B-fragment register image (f16 pairs packed per dword), so each lane loads
// a fragment as 32 contiguous bytes (2x ds_load_b128). All 8 B fragments
// (2 K-halves x 4 N-tiles) stay register-resident while each wave streams
// TILES_PER_WAVE 16-row tiles of A through the WMMA pipe.
// ---------------------------------------------------------------------------
__global__ void gemm_bias_relu_wmma(const float* __restrict__ X,
                                    const float* __restrict__ W,
                                    const float* __restrict__ bias,
                                    float* __restrict__ out, int nRows) {
    // Bl[kh][t][lane][j] : packed {W[k][col], W[k+1][col]} f16x2
    //   k = kh*32 + (lane&16 ? 16 : 0) + 2j ; col = t*16 + (lane&15)
    __shared__ __align__(32) unsigned int Bl[2048];   // 8 KB
    const int tid = threadIdx.x;
    for (int idx = tid; idx < 2048; idx += 128) {
        const int kh   = idx >> 10;
        const int t    = (idx >> 8) & 3;
        const int ln   = (idx >> 3) & 31;
        const int j    = idx & 7;
        const int k    = kh * 32 + ((ln & 16) ? 16 : 0) + 2 * j;
        const int col  = t * 16 + (ln & 15);
        union { _Float16 h[2]; unsigned int u; } p;
        p.h[0] = (_Float16)W[k * FDIM + col];
        p.h[1] = (_Float16)W[(k + 1) * FDIM + col];
        Bl[idx] = p.u;
    }
    __syncthreads();

    const int wave = tid >> 5;
    const int lane = tid & 31;
    const int halfLane = lane & 15;

    // B fragments -> registers (wide LDS loads, once per wave)
    v16h bf[2][4];
#pragma unroll
    for (int kh = 0; kh < 2; ++kh)
#pragma unroll
        for (int t = 0; t < 4; ++t)
            bf[kh][t] = *(const v16h*)&Bl[(kh * 4 + t) * 256 + lane * 8];

    // A fragment, contiguous-k order used here (matches Bl): lanes 0-15 hold K {0..15},
    // lanes 16-31 hold K {16..31} of row M = lane%16
    const int kLo  = (lane < 16) ? 0 : 16;
    const int kHi  = (lane < 16) ? 8 : 24;
    const int mOff = (lane < 16) ? 0 : 8;

    const int tileBase = (blockIdx.x * 4 + wave) * TILES_PER_WAVE;
#pragma unroll
    for (int tt = 0; tt < TILES_PER_WAVE; ++tt) {
        const int rowBase = (tileBase + tt) * 16;
        if (rowBase >= nRows) return;              // wave-uniform guard
        const float* __restrict__ arow = X + (size_t)(rowBase + halfLane) * FDIM;

        v8f acc[4];
#pragma unroll
        for (int t = 0; t < 4; ++t) acc[t] = (v8f){0.f,0.f,0.f,0.f,0.f,0.f,0.f,0.f};

#pragma unroll
        for (int k0 = 0; k0 < FDIM; k0 += 32) {
            v16h a;
#pragma unroll
            for (int j = 0; j < 8; ++j) a[j]     = (_Float16)arow[k0 + kLo + j];
#pragma unroll
            for (int j = 0; j < 8; ++j) a[8 + j] = (_Float16)arow[k0 + kHi + j];
#pragma unroll
            for (int t = 0; t < 4; ++t)
                acc[t] = __builtin_amdgcn_wmma_f32_16x16x32_f16(
                    false, a, false, bf[k0 >> 5][t], (short)0, acc[t], false, false);
        }

        // C/D layout: VGPR r -> M=r (lanes 0-15) / M=8+r (lanes 16-31); N=lane%16
#pragma unroll
        for (int t = 0; t < 4; ++t) {
            const int col = t * 16 + halfLane;
            const float bv = bias[col];
#pragma unroll
            for (int r = 0; r < 8; ++r) {
                float v = acc[t][r] + bv;
                v = v > 0.0f ? v : 0.0f;
                out[(size_t)(rowBase + mOff + r) * FDIM + col] = v;
            }
        }
    }
}

// ---------------------------------------------------------------------------
// edge scatter: msg[dst] += z[src]   (one wave per edge, float2 per lane)
// wave reads a full 256B source row coalesced, atomics resolve in L2
// ---------------------------------------------------------------------------
__global__ void scatter_kernel(const float* __restrict__ z,
                               const int* __restrict__ src, const int* __restrict__ dst,
                               float* __restrict__ msg, int E) {
    long long gid = (long long)blockIdx.x * blockDim.x + threadIdx.x;
    long long e = gid >> 5;
    if (e >= E) return;
    const int lane = (int)(gid & 31);
    const int s = src[e], d = dst[e];
    const float2 v = ((const float2*)(z + (size_t)s * FDIM))[lane];
    float* mp = msg + (size_t)d * FDIM + lane * 2;
    atomicAdd(mp,     v.x);
    atomicAdd(mp + 1, v.y);
}

// ---------------------------------------------------------------------------
// combine: out = z - ((msg + z)/cnt)/deg ; re-zeros msg for the next filter
// (self-loop contribution z[i] folded in here instead of via atomics)
// ---------------------------------------------------------------------------
__global__ void combine_kernel(const float* __restrict__ z, float* __restrict__ msg,
                               const float* __restrict__ cnt, const float* __restrict__ deg,
                               float* __restrict__ out, int n) {
    long long i = (long long)blockIdx.x * blockDim.x + threadIdx.x;
    if (i >= (long long)n * FDIM) return;
    const int node = (int)(i >> 6);
    const float zi = z[i];
    const float m  = msg[i] + zi;
    msg[i] = 0.0f;
    out[i] = zi - (m / cnt[node]) / deg[node];
}

// ---------------------------------------------------------------------------
extern "C" void kernel_launch(void* const* d_in, const int* in_sizes, int n_in,
                              void* d_out, int out_size, void* d_ws, size_t ws_size,
                              hipStream_t stream) {
    const float* x    = (const float*)d_in[0];
    const int*   edge = (const int*)  d_in[1];
    const float* W1   = (const float*)d_in[2];
    const float* b1   = (const float*)d_in[3];
    const float* W2   = (const float*)d_in[4];
    const float* b2   = (const float*)d_in[5];
    const float* W3   = (const float*)d_in[6];
    const float* b3   = (const float*)d_in[7];
    float* out = (float*)d_out;

    const int n = in_sizes[0] / FDIM;       // 100000 nodes
    const int E = in_sizes[1] / 2;          // 1200000 edges
    const int* src = edge;
    const int* dst = edge + E;

    // workspace layout
    float* zA  = (float*)d_ws;
    float* zB  = zA  + (size_t)n * FDIM;
    float* msg = zB  + (size_t)n * FDIM;
    float* deg = msg + (size_t)n * FDIM;
    float* cnt = deg + n;

    const long long tot = (long long)n * FDIM;
    const int elemBlocks = (int)((tot + 255) / 256);
    const int tiles      = (n + 15) / 16;
    const int wavesNeed  = (tiles + TILES_PER_WAVE - 1) / TILES_PER_WAVE;
    const int gemmBlocks = (wavesNeed + 3) / 4;                 // 4 waves/block
    const int scatBlocks = (int)(((long long)E * 32 + 255) / 256);

    init_kernel<<<elemBlocks, 256, 0, stream>>>(deg, cnt, msg, n);
    deg_cnt_kernel<<<(E + 255) / 256, 256, 0, stream>>>(src, dst, deg, cnt, E);

    // layer 1
    gemm_bias_relu_wmma<<<gemmBlocks, 128, 0, stream>>>(x, W1, b1, zA, n);
    scatter_kernel<<<scatBlocks, 256, 0, stream>>>(zA, src, dst, msg, E);
    combine_kernel<<<elemBlocks, 256, 0, stream>>>(zA, msg, cnt, deg, zB, n);

    // layer 2
    gemm_bias_relu_wmma<<<gemmBlocks, 128, 0, stream>>>(zB, W2, b2, zA, n);
    scatter_kernel<<<scatBlocks, 256, 0, stream>>>(zA, src, dst, msg, E);
    combine_kernel<<<elemBlocks, 256, 0, stream>>>(zA, msg, cnt, deg, zB, n);

    // layer 3 -> output
    gemm_bias_relu_wmma<<<gemmBlocks, 128, 0, stream>>>(zB, W3, b3, out, n);
}
